// Attention_23871428231614
// MI455X (gfx1250) — matrix operand order repaired and checked by reading, not compiled, not run on hardware
//
#include <hip/hip_runtime.h>
#include <hip/hip_bf16.h>
#include <stdint.h>

// ---------------------------------------------------------------------------
// Sizes (b=1 fixed by the reference)
// ---------------------------------------------------------------------------
#define NSEQ    2048
#define DIMM    1024
#define NHEADS  8
#define DQK     128
#define DVV     192
#define DPW     128
#define NPW     512
#define QKV_OUT 1344          // 8*128 + 128 + 192
#define HDV     (NHEADS*DVV)  // 1536

typedef __attribute__((ext_vector_type(8)))  float  v8f;
typedef __attribute__((ext_vector_type(16))) __bf16 v16bf;
typedef __attribute__((ext_vector_type(8)))  __bf16 bf8;

#define DEV __device__ __forceinline__

DEV v16bf cat16(bf8 lo, bf8 hi) {
  v16bf r;
#pragma unroll
  for (int i = 0; i < 8; ++i) { r[i] = lo[i]; r[i + 8] = hi[i]; }
  return r;
}

// A operand (16x32 bf16): lane<16 -> K[0..7]+K[16..23]; lane>=16 -> K[8..15]+K[24..31]
DEV v16bf load_a32(const __bf16* row, int lane) {
  int s = (lane >> 4) * 8;
  bf8 c0 = *(const bf8*)(row + s);
  bf8 c1 = *(const bf8*)(row + 16 + s);
  return cat16(c0, c1);
}
// B operand (32x16 bf16): lane<16 -> K[0..7]+K[16..23]; lane>=16 -> K[8..15]+K[24..31]
DEV v16bf load_b32(const __bf16* row, int lane) {
  int s = (lane >> 4) * 8;
  bf8 c0 = *(const bf8*)(row + s);
  bf8 c1 = *(const bf8*)(row + 16 + s);
  return cat16(c0, c1);
}
DEV v8f wmma_bf16(v16bf a, v16bf b, v8f c) {
  return __builtin_amdgcn_wmma_f32_16x16x32_bf16(false, a, false, b, (short)0, c,
                                                 false, false);
}

// Async memory->LDS b128 copy (ASYNCcnt path), GVS addressing: base(SGPR64)+voff(VGPR32).
DEV void async_b128(unsigned lds_addr, unsigned voff, unsigned long long base) {
  asm volatile("global_load_async_to_lds_b128 %0, %1, %2"
               :: "v"(lds_addr), "v"(voff), "s"(base)
               : "memory");
}
DEV void wait_async_le10() { asm volatile("s_wait_asynccnt 0xa" ::: "memory"); }
DEV void wait_async_0()    { asm volatile("s_wait_asynccnt 0x0" ::: "memory"); }

// ---------------------------------------------------------------------------
// Prep kernels
// ---------------------------------------------------------------------------
__global__ void cvt_bf16_kernel(const float* __restrict__ s, __bf16* __restrict__ d, int n) {
  int i = blockIdx.x * 256 + threadIdx.x;
  if (i < n) d[i] = (__bf16)s[i];
}

// dst[c*R + r] = src[r*C + c]   (f32 -> bf16 transpose)
__global__ void transpose_bf16_kernel(const float* __restrict__ s, __bf16* __restrict__ d,
                                      int R, int C) {
  int i = blockIdx.x * 256 + threadIdx.x;
  if (i < R * C) {
    int r = i / C, c = i % C;
    d[(size_t)c * R + r] = (__bf16)s[i];
  }
}

// ---------------------------------------------------------------------------
// Generic bf16 WMMA GEMM: C[M,N] = A[M,K] * B[K,N], B given transposed BT[N,K].
// Block = 8 waves; each wave computes a 16x64 C tile.
// ---------------------------------------------------------------------------
__global__ void __launch_bounds__(256) gemm_bf16_kernel(
    const __bf16* __restrict__ A, const __bf16* __restrict__ BT,
    float* __restrict__ C, int M, int N, int K) {
  const int lane = threadIdx.x & 31, wave = threadIdx.x >> 5;
  const int m0 = (blockIdx.x * 8 + wave) * 16;
  const int n0 = blockIdx.y * 64;
  v8f acc[4];
#pragma unroll
  for (int j = 0; j < 4; ++j)
#pragma unroll
    for (int i = 0; i < 8; ++i) acc[j][i] = 0.f;

  const __bf16* arow  = A  + (size_t)(m0 + (lane & 15)) * K;
  const __bf16* brow0 = BT + (size_t)(n0 + (lane & 15)) * K;

  for (int k = 0; k < K; k += 32) {
    v16bf a = load_a32(arow + k, lane);
#pragma unroll
    for (int j = 0; j < 4; ++j) {
      v16bf b = load_b32(brow0 + (size_t)(j * 16) * K + k, lane);
      acc[j] = wmma_bf16(a, b, acc[j]);
    }
  }
  const int mb = (lane >> 4) * 8, nn = lane & 15;
#pragma unroll
  for (int j = 0; j < 4; ++j)
#pragma unroll
    for (int rr = 0; rr < 8; ++rr)
      C[(size_t)(m0 + mb + rr) * N + n0 + j * 16 + nn] = acc[j][rr];
}

// ---------------------------------------------------------------------------
// Post-QKV: RMSNorm(q,k,v) + scale + RoPE.
// Emits bf16 q[8][n][128], k[n][128] (row-major) and V TRANSPOSED vT[192][n]
// so the attention kernel can fetch V tiles with pure b128 async copies.
// ---------------------------------------------------------------------------
__global__ void __launch_bounds__(256) postprocess_kernel(
    const float* __restrict__ qkv, const float* __restrict__ rot,
    const float* __restrict__ qn, const float* __restrict__ kn,
    const float* __restrict__ vn,
    __bf16* __restrict__ qo, __bf16* __restrict__ ko, __bf16* __restrict__ vtg) {
  const int row = blockIdx.x, tid = threadIdx.x;
  const int lane = tid & 31, wave = tid >> 5;
  __shared__ float r[QKV_OUT];
  __shared__ float inv[10];
  const float* src = qkv + (size_t)row * QKV_OUT;
  for (int i = tid; i < QKV_OUT; i += 256) r[i] = src[i];
  __syncthreads();
  for (int pass = 0; pass < 2; ++pass) {
    int seg = wave + pass * 8;
    if (seg < 10) {
      int off = (seg < 8) ? seg * DQK : (seg == 8 ? 1024 : 1152);
      int len = (seg == 9) ? DVV : DQK;
      float s = 0.f;
      for (int i = lane; i < len; i += 32) { float t = r[off + i]; s += t * t; }
      for (int m = 16; m; m >>= 1) s += __shfl_xor(s, m, 32);
      if (lane == 0) inv[seg] = rsqrtf(s / (float)len + 1.1920929e-07f);
    }
  }
  __syncthreads();
  const float* rp = rot + (size_t)row * DQK;
  for (int e = tid; e < NHEADS * DQK; e += 256) {   // q heads, scaled 64^-0.5
    int h = e >> 7, j = e & 127;
    float sc = inv[h] * 0.125f;
    float t = r[h * DQK + j] * sc * qn[j];
    int j2 = (j < 64) ? j + 64 : j - 64;
    float rh = r[h * DQK + j2] * sc * qn[j2];
    if (j < 64) rh = -rh;
    float p = rp[j];
    qo[((size_t)h * NSEQ + row) * DQK + j] = (__bf16)(t * __cosf(p) + rh * __sinf(p));
  }
  for (int j = tid; j < DQK; j += 256) {            // k
    float t = r[1024 + j] * inv[8] * kn[j];
    int j2 = (j < 64) ? j + 64 : j - 64;
    float rh = r[1024 + j2] * inv[8] * kn[j2];
    if (j < 64) rh = -rh;
    float p = rp[j];
    ko[(size_t)row * DQK + j] = (__bf16)(t * __cosf(p) + rh * __sinf(p));
  }
  for (int j = tid; j < DVV; j += 256)              // v, transposed [dv][n]
    vtg[(size_t)j * NSEQ + row] = (__bf16)(r[1152 + j] * inv[9] * vn[j]);
}

// ---------------------------------------------------------------------------
// Pairwise bias: BN(eval) -> exact GELU -> x@W_bias -> bias[8][512][512]
// Streams the 134MB pairwise tensor exactly once (the bandwidth floor).
// ---------------------------------------------------------------------------
__global__ void __launch_bounds__(256) bias_kernel(
    const float* __restrict__ pw, const float* __restrict__ gamma,
    const float* __restrict__ beta, const float* __restrict__ rv,
    const float* __restrict__ wb, float* __restrict__ bias) {
  __shared__ float s_wb[DPW][NHEADS];
  __shared__ float s_scale[DPW], s_beta[DPW];
  const int tid = threadIdx.x, lane = tid & 31, wave = tid >> 5;
  for (int i = tid; i < DPW * NHEADS; i += 256) s_wb[i / NHEADS][i % NHEADS] = wb[i];
  for (int c = tid; c < DPW; c += 256) {
    float stdv = sqrtf(fmaxf(rv[c], 1e-5f));
    s_scale[c] = 11.313708498984761f / stdv * (gamma[c] + 1.0f);  // sqrt(128)/std*(g+1)
    s_beta[c] = beta[c];
  }
  __syncthreads();
  for (int t = 0; t < 8; ++t) {
    int p = blockIdx.x * 64 + wave * 8 + t;
    __builtin_prefetch(pw + (size_t)(p + 1) * DPW + lane * 4, 0, 1);  // global_prefetch
    float4 x4 = *(const float4*)(pw + (size_t)p * DPW + lane * 4);
    float acc[NHEADS];
#pragma unroll
    for (int h = 0; h < 8; ++h) acc[h] = 0.f;
    float xv[4] = {x4.x, x4.y, x4.z, x4.w};
#pragma unroll
    for (int q = 0; q < 4; ++q) {
      int c = lane * 4 + q;
      float x = xv[q] * s_scale[c] + s_beta[c];
      float g = 0.5f * x * (1.0f + erff(x * 0.7071067811865475f));
#pragma unroll
      for (int h = 0; h < 8; ++h) acc[h] += g * s_wb[c][h];
    }
    for (int m = 16; m; m >>= 1)
#pragma unroll
      for (int h = 0; h < 8; ++h) acc[h] += __shfl_xor(acc[h], m, 32);
    if (lane < 8) bias[(size_t)lane * (NPW * NPW) + p] = acc[lane];
  }
}

// ---------------------------------------------------------------------------
// Flash attention, double-buffered async-to-LDS tile pipeline.
//  - soft-clamp bounds logits to [-5,5] => no running max needed
//  - ones tile (sOnes) makes the softmax denominator WMMA tile #12
//  - K tile (16KB) and Vt tile (24KB) arrive via global_load_async_to_lds_b128
//    (10 issues/thread); prefetch of tile i+1 overlaps WMMA on tile i, gated by
//    s_wait_asynccnt (in-order completion => wait<=10 completes current tile).
// Block = 8 waves (128 q rows); 42 wmma per wave per 64-key tile.
// ---------------------------------------------------------------------------
__global__ void __launch_bounds__(256) attn_kernel(
    const __bf16* __restrict__ q, const __bf16* __restrict__ kmat,
    const __bf16* __restrict__ vtg, const float* __restrict__ bias,
    __bf16* __restrict__ out) {
  __shared__ __bf16 sK[2][64][DQK];     // 2 x 16 KB
  __shared__ __bf16 sVt[2][DVV][64];    // 2 x 24 KB
  __shared__ __bf16 sOnes[16][64];      // 2 KB  (row 0 = ones -> rowsum column)
  __shared__ __bf16 sP[8][16][64];      // 16 KB, wave-private P staging
  const int tid = threadIdx.x, lane = tid & 31, wave = tid >> 5;
  const int head = blockIdx.y;
  const int qrow0 = blockIdx.x * 128 + wave * 16;
  const int mb = (lane >> 4) * 8, nn = lane & 15;
  const unsigned long long kbase = (unsigned long long)(uintptr_t)kmat;
  const unsigned long long vbase = (unsigned long long)(uintptr_t)vtg;

  for (int i = tid; i < 16 * 64; i += 256)
    sOnes[i >> 6][i & 63] = (i < 64) ? (__bf16)1.0f : (__bf16)0.0f;

  // Q A-operands live in registers for the whole pass (16x128 per wave)
  const __bf16* qrow = q + ((size_t)head * NSEQ + qrow0 + (lane & 15)) * DQK;
  v16bf qa[4];
#pragma unroll
  for (int c = 0; c < 4; ++c) qa[c] = load_a32(qrow + c * 32, lane);

  v8f accO[13];
#pragma unroll
  for (int t = 0; t < 13; ++t)
#pragma unroll
    for (int i = 0; i < 8; ++i) accO[t][i] = 0.f;

  // issue tile 0 into buffer 0: K = 1024 b128 chunks, Vt = 1536 b128 chunks
  {
    for (int i = tid; i < 1024; i += 256) {
      int rr = i >> 4, cc = (i & 15) * 8;
      async_b128((unsigned)(uintptr_t)&sK[0][rr][cc],
                 (unsigned)((rr * DQK + cc) * 2), kbase);
    }
    for (int i = tid; i < 1536; i += 256) {
      int rr = i >> 3, cc = (i & 7) * 8;
      async_b128((unsigned)(uintptr_t)&sVt[0][rr][cc],
                 (unsigned)((rr * NSEQ + cc) * 2), vbase);
    }
  }

  for (int kt = 0; kt < NSEQ; kt += 64) {
    const int buf = (kt >> 6) & 1;
    if (kt + 64 < NSEQ) {               // prefetch next tile into other buffer
      const int nb = buf ^ 1, nkt = kt + 64;
      for (int i = tid; i < 1024; i += 256) {
        int rr = i >> 4, cc = (i & 15) * 8;
        async_b128((unsigned)(uintptr_t)&sK[nb][rr][cc],
                   (unsigned)(((nkt + rr) * DQK + cc) * 2), kbase);
      }
      for (int i = tid; i < 1536; i += 256) {
        int rr = i >> 3, cc = (i & 7) * 8;
        async_b128((unsigned)(uintptr_t)&sVt[nb][rr][cc],
                   (unsigned)((rr * NSEQ + nkt + cc) * 2), vbase);
      }
      wait_async_le10();                // current tile complete, prefetch in flight
    } else {
      wait_async_0();
    }
    __syncthreads();

    // S = Q K^T : 4 key sub-tiles x 4 k-chunks
    v8f s[4];
#pragma unroll
    for (int j = 0; j < 4; ++j)
#pragma unroll
      for (int i = 0; i < 8; ++i) s[j][i] = 0.f;
#pragma unroll
    for (int j = 0; j < 4; ++j) {
      const __bf16* krow = &sK[buf][j * 16 + nn][0];
#pragma unroll
      for (int c = 0; c < 4; ++c)
        s[j] = wmma_bf16(qa[c], load_b32(krow + c * 32, lane), s[j]);
    }
    // bias + tanh softclamp + exp -> P (bf16) in wave-private LDS
#pragma unroll
    for (int j = 0; j < 4; ++j) {
      int keyg = kt + j * 16 + nn;
#pragma unroll
      for (int rr = 0; rr < 8; ++rr) {
        int qg = qrow0 + mb + rr;
        float bv = bias[((size_t)head * NPW + (qg >> 2)) * NPW + (keyg >> 2)];
        float tt = tanhf((s[j][rr] + bv) * 0.2f) * 5.0f;
        sP[wave][mb + rr][j * 16 + nn] = (__bf16)__expf(tt);
      }
    }
    // O += P V  (tile 12 = ones column => softmax denominator)
#pragma unroll
    for (int c = 0; c < 2; ++c) {
      v16bf pa = load_a32(&sP[wave][lane & 15][0] + c * 32, lane);
#pragma unroll
      for (int t = 0; t < 12; ++t)
        accO[t] = wmma_bf16(pa, load_b32(&sVt[buf][t * 16 + nn][0] + c * 32, lane),
                            accO[t]);
      accO[12] = wmma_bf16(pa, load_b32(&sOnes[nn][0] + c * 32, lane), accO[12]);
    }
    __syncthreads();                    // done reading buf before it is re-filled
  }
  // epilogue: divide by denominator (column 0 of tile 12), write [n][h*192+dv]
#pragma unroll
  for (int rr = 0; rr < 8; ++rr) {
    float den = __shfl(accO[12][rr], lane & 16, 32);
    float id = 1.0f / den;
    int row = qrow0 + mb + rr;
#pragma unroll
    for (int t = 0; t < 12; ++t)
      out[(size_t)row * HDV + head * DVV + t * 16 + nn] = (__bf16)(accO[t][rr] * id);
  }
}

// ---------------------------------------------------------------------------
// Launch
// ---------------------------------------------------------------------------
static inline size_t alignup(size_t x) { return (x + 255) & ~(size_t)255; }

extern "C" void kernel_launch(void* const* d_in, const int* in_sizes, int n_in,
                              void* d_out, int out_size, void* d_ws, size_t ws_size,
                              hipStream_t stream) {
  const float* x    = (const float*)d_in[0];
  const float* pw   = (const float*)d_in[1];
  const float* rot  = (const float*)d_in[2];
  const float* Wqkv = (const float*)d_in[3];
  const float* qn   = (const float*)d_in[4];
  const float* kn   = (const float*)d_in[5];
  const float* vn   = (const float*)d_in[6];
  const float* bg   = (const float*)d_in[7];
  const float* bb   = (const float*)d_in[8];
  const float* brv  = (const float*)d_in[9];
  const float* Wb   = (const float*)d_in[10];
  const float* Wout = (const float*)d_in[11];
  float* out = (float*)d_out;

  char* ws = (char*)d_ws;
  size_t off = 0;
  __bf16* xb    = (__bf16*)(ws + off); off += alignup((size_t)NSEQ * DIMM * 2);
  __bf16* wqkvT = (__bf16*)(ws + off); off += alignup((size_t)QKV_OUT * DIMM * 2);
  __bf16* woutT = (__bf16*)(ws + off); off += alignup((size_t)DIMM * HDV * 2);
  float*  qkv   = (float*) (ws + off); off += alignup((size_t)NSEQ * QKV_OUT * 4);
  __bf16* qb    = (__bf16*)(ws + off); off += alignup((size_t)NHEADS * NSEQ * DQK * 2);
  __bf16* kb    = (__bf16*)(ws + off); off += alignup((size_t)NSEQ * DQK * 2);
  __bf16* vtg   = (__bf16*)(ws + off); off += alignup((size_t)DVV * NSEQ * 2);
  float*  biasb = (float*) (ws + off); off += alignup((size_t)NHEADS * NPW * NPW * 4);
  __bf16* ao    = (__bf16*)(ws + off); off += alignup((size_t)NSEQ * HDV * 2);
  (void)off; (void)ws_size; (void)in_sizes; (void)n_in; (void)out_size;

  cvt_bf16_kernel<<<(NSEQ * DIMM + 255) / 256, 256, 0, stream>>>(x, xb, NSEQ * DIMM);
  transpose_bf16_kernel<<<(DIMM * QKV_OUT + 255) / 256, 256, 0, stream>>>(Wqkv, wqkvT, DIMM, QKV_OUT);
  transpose_bf16_kernel<<<(HDV * DIMM + 255) / 256, 256, 0, stream>>>(Wout, woutT, HDV, DIMM);

  gemm_bf16_kernel<<<dim3(NSEQ / 128, QKV_OUT / 64), 256, 0, stream>>>(
      xb, wqkvT, qkv, NSEQ, QKV_OUT, DIMM);

  postprocess_kernel<<<NSEQ, 256, 0, stream>>>(qkv, rot, qn, kn, vn, qb, kb, vtg);

  bias_kernel<<<(NPW * NPW) / 64, 256, 0, stream>>>(pw, bg, bb, brv, Wb, biasb);

  attn_kernel<<<dim3(NSEQ / 128, NHEADS), 256, 0, stream>>>(qb, kb, vtg, biasb, ao);

  gemm_bf16_kernel<<<dim3(NSEQ / 128, DIMM / 64), 256, 0, stream>>>(
      ao, woutT, out, NSEQ, DIMM, HDV);
}
